// LWRLoss_71296457113982
// MI455X (gfx1250) — matrix operand order repaired and checked by reading, not compiled, not run on hardware
//
#include <hip/hip_runtime.h>
#include <hip/hip_bf16.h>

typedef __attribute__((ext_vector_type(16))) _Float16 v16h;
typedef __attribute__((ext_vector_type(8)))  float    v8f;
typedef unsigned int v4u __attribute__((ext_vector_type(4)));
typedef int          v8i __attribute__((ext_vector_type(8)));
typedef int          v4i __attribute__((ext_vector_type(4)));

#define BQ      1024
#define DIM     64
#define NPTS    100000
#define KNN     100
#define PDIM    65                  // DIM + 1
#define CHUNK   2048
#define SUBC    256
#define NSUB    (CHUNK / SUBC)      // 8 sub-chunks per chunk
#define NCHUNK  49                  // ceil(100000 / 2048)
#define NPAD    (NCHUNK * CHUNK)    // 100352 padded DB rows
#define CPR     (NCHUNK * KNN)      // 4900 candidates per query row
#define JITTER  1e-6f

__device__ __forceinline__ unsigned orderKeyBits(unsigned b) {
  return b ^ ((b & 0x80000000u) ? 0xFFFFFFFFu : 0x80000000u);
}

// ---------------------------------------------------------------------------
// TDM tile load: 256 rows x 128 bytes (64 f16) from global -> LDS.
// D# per CDNA5 ISA:
//   group0: count=1 | lds_addr | global_addr[56:0] | type=2
//   group1: data_size=3 (8B), tensor 16x256 (u64 elems), tile 16x256,
//           dim0_stride=16, dim1_stride=16 (contiguous rows)
// Source is pre-padded, so tensor == tile (no OOB path).
// This toolchain's builtin takes 6 args (g0, g1, g2, g3, extra group, cpol).
// ---------------------------------------------------------------------------
__device__ __forceinline__ void tdm_load_tile(unsigned lds_off,
                                              unsigned long long gaddr) {
  v4u g0;
  g0[0] = 1u;                                                   // count=1
  g0[1] = lds_off;                                              // lds_addr
  g0[2] = (unsigned)(gaddr & 0xFFFFFFFFull);                    // ga[31:0]
  g0[3] = ((unsigned)(gaddr >> 32) & 0x01FFFFFFu) | 0x80000000u; // ga[56:32]|type=2
  v8i g1;
  g1[0] = 0x00030000;        // workgroup_mask=0, data_size=3 (8 bytes)
  g1[1] = 16 << 16;          // tensor_dim0 = 16 (u64 elems per row)
  g1[2] = 256 << 16;         // tensor_dim1 = 256 (rows), low half of field
  g1[3] = 16 << 16;          // tensor_dim1 hi = 0 | tile_dim0 = 16
  g1[4] = 256;               // tile_dim1 = 256, tile_dim2 = 0
  g1[5] = 16;                // tensor_dim0_stride = 16 elems (128 B)
  g1[6] = 16 << 16;          // tensor_dim1_stride = 16 elems
  g1[7] = 0;
  v4i z4 = {0, 0, 0, 0};
  v8i z8 = {0, 0, 0, 0, 0, 0, 0, 0};
  __builtin_amdgcn_tensor_load_to_lds(g0, g1, z4, z4, z8, 0);
}

// ---------------------------------------------------------------------------
// Kernel P: one-time prep. db_X -> f16 copy (L2-resident, 12.8 MB) + row
// norms; pad rows [NPTS, NPAD) with zeros / +inf so the hot loop is branch-free.
// ---------------------------------------------------------------------------
__global__ void __launch_bounds__(256)
prep_db_kernel(const float* __restrict__ dbX,
               _Float16* __restrict__ dbXh,
               float* __restrict__ d2)
{
  const int n = blockIdx.x * blockDim.x + threadIdx.x;
  if (n >= NPAD) return;
  if (n < NPTS) {
    float s = 0.f;
    #pragma unroll
    for (int k = 0; k < DIM; ++k) {
      const float v = dbX[(size_t)n * DIM + k];
      s += v * v;
      dbXh[(size_t)n * DIM + k] = (_Float16)v;
    }
    d2[n] = s;
  } else {
    #pragma unroll
    for (int k = 0; k < DIM; ++k) dbXh[(size_t)n * DIM + k] = (_Float16)0.f;
    d2[n] = __builtin_inff();
  }
}

// ---------------------------------------------------------------------------
// Kernel A: fused distance GEMM (WMMA f16->f32) + per-chunk exact top-K.
// DB tiles staged by the Tensor Data Mover, double-buffered against WMMA.
// grid = (NCHUNK, BQ/16), block = 256 (8 wave32).
// Dynamic LDS layout (bytes):
//   [0      ) AqF16   16*64 f16        = 2048
//   [2048   ) x2      16 f32           = 64
//   [2112   ) dbF16   2 x 256*64 f16   = 65536
//   [67648  ) dist    16*2048 f32      = 131072   (aliased as uint keys)
//   [198720 ) rowCnt  16 int           = 64       total = 198784
// ---------------------------------------------------------------------------
__global__ void __launch_bounds__(256)
dist_topk_chunk_kernel(const float* __restrict__ input,
                       const _Float16* __restrict__ dbXh,
                       const float* __restrict__ d2g,
                       unsigned* __restrict__ candKey,
                       int* __restrict__ candIdx)
{
  extern __shared__ char smem[];
  _Float16* AqF16 = (_Float16*)(smem);
  float*    x2    = (float*)(smem + 2048);
  _Float16* dbF16 = (_Float16*)(smem + 2112);      // two 32 KB buffers
  float*    dist  = (float*)(smem + 67648);
  unsigned* keys  = (unsigned*)(smem + 67648);
  int*      rowCnt= (int*)(smem + 198720);

  const int tid  = threadIdx.x;
  const int lane = tid & 31;
  const int wave = tid >> 5;
  const int hi   = lane >> 4;
  const int nl   = lane & 15;
  const int chunk  = blockIdx.x;
  const int rowBlk = blockIdx.y;
  const int n0 = chunk * CHUNK;

  const unsigned ldsBuf0 = (unsigned)(size_t)(void*)dbF16;
  const unsigned long long dbBase = (unsigned long long)(size_t)dbXh;

  // Stage 16 query rows: fp32 norms + f16 copy for the WMMA A operand.
  if (tid < 16) {
    const int gr = rowBlk * 16 + tid;
    float s = 0.f;
    #pragma unroll
    for (int k = 0; k < DIM; ++k) {
      const float v = input[(size_t)gr * DIM + k];
      s += v * v;
      AqF16[tid * DIM + k] = (_Float16)v;
    }
    x2[tid] = s;
    rowCnt[tid] = 0;
  }

  // Preload sub-chunk 0 via TDM (wave 0 issues; EXEC ignored by tensor ops).
  if (wave == 0)
    tdm_load_tile(ldsBuf0, dbBase + (unsigned long long)n0 * (DIM * 2));
  __builtin_amdgcn_s_wait_tensorcnt((short)0);
  __syncthreads();

  // Per-lane A fragments (fixed for the whole chunk).
  // A 16x32 f16 layout: M = lane&15, K(e) = (e&7) + 16*(e>>3) + 8*(lane>>4)
  v16h afrag[2];
  #pragma unroll
  for (int t = 0; t < 2; ++t) {
    #pragma unroll
    for (int e = 0; e < 16; ++e) {
      const int k = t * 32 + (e & 7) + ((e >> 3) << 4) + 8 * hi;
      afrag[t][e] = AqF16[nl * DIM + k];
    }
  }

  for (int sc = 0; sc < NSUB; ++sc) {
    // Issue DMA for the next tile into the other buffer while we compute.
    if (wave == 0 && sc + 1 < NSUB)
      tdm_load_tile(ldsBuf0 + (unsigned)(((sc + 1) & 1) * SUBC * DIM * 2),
                    dbBase + (unsigned long long)(n0 + (sc + 1) * SUBC) * (DIM * 2));

    const _Float16* dbB = dbF16 + (sc & 1) * (SUBC * DIM);
    const int base = n0 + sc * SUBC;

    // 16 N-tiles of 16x16 per sub-chunk; 2 per wave; K=64 -> 2 WMMA each.
    #pragma unroll
    for (int tt = 0; tt < 2; ++tt) {
      const int nt   = wave * 2 + tt;
      const int nloc = nt * 16 + nl;      // B/C/D operand: N = lane&15
      v8f acc = {};
      #pragma unroll
      for (int t = 0; t < 2; ++t) {
        v16h bf;
        #pragma unroll
        for (int e = 0; e < 16; ++e)      // B 32x16: K(e) = (e&7) + 16*(e>>3) + 8*(lane>>4)
          bf[e] = dbB[nloc * DIM + t * 32 + (e & 7) + ((e >> 3) << 4) + 8 * hi];
        acc = __builtin_amdgcn_wmma_f32_16x16x32_f16(false, afrag[t], false, bf,
                                                     (short)0, acc, false, false);
      }
      const float dd2 = d2g[base + nloc];   // L2-resident, padded with +inf
      #pragma unroll
      for (int r = 0; r < 8; ++r) {         // C/D: M = r + 8*(lane>>4)
        const int m = r + 8 * hi;
        dist[m * CHUNK + sc * SUBC + nloc] =
            (x2[m] - 2.0f * acc[r] + dd2) * (1.0f / (float)DIM);
      }
    }

    __builtin_amdgcn_s_wait_tensorcnt((short)0);  // next tile landed
    __syncthreads();                              // all waves see it
  }

  // In-place convert distances to order-preserving uint keys (pure bit ops).
  for (int i = tid; i < 16 * CHUNK; i += 256) keys[i] = orderKeyBits(keys[i]);
  __syncthreads();

  // Per-wave exact top-K via 32-step bit bisection; wave handles 2 rows.
  for (int rr = 0; rr < 2; ++rr) {
    const int m = wave * 2 + rr;
    const unsigned* krow = keys + m * CHUNK;

    unsigned prefix = 0;                  // max T with count(key < T) <= K-1
    for (int bit = 31; bit >= 0; --bit) {
      const unsigned trial = prefix | (1u << bit);
      int c = 0;
      for (int i = lane; i < CHUNK; i += 32) c += (krow[i] < trial) ? 1 : 0;
      #pragma unroll
      for (int o = 16; o > 0; o >>= 1) c += __shfl_xor(c, o);
      if (c <= KNN - 1) prefix = trial;
    }
    const unsigned T = prefix;            // value of the K-th smallest key

    const int gr = rowBlk * 16 + m;
    unsigned* ck = candKey + (size_t)gr * CPR + (size_t)chunk * KNN;
    int*      ci = candIdx + (size_t)gr * CPR + (size_t)chunk * KNN;
    for (int i = lane; i < CHUNK; i += 32) {
      if (krow[i] < T) {
        const int pos = atomicAdd(&rowCnt[m], 1);
        ck[pos] = krow[i]; ci[pos] = n0 + i;
      }
    }
    for (int i = lane; i < CHUNK; i += 32) {   // fill remaining slots with ties
      if (krow[i] == T) {
        const int pos = atomicAdd(&rowCnt[m], 1);
        if (pos < KNN) { ck[pos] = krow[i]; ci[pos] = n0 + i; }
      }
    }
  }
}

// ---------------------------------------------------------------------------
// Kernel B: merge candidates -> final top-K, gather neighbors, solve the
// 65x65 ridge system per row, predict, accumulate MSE. One block per row.
// ---------------------------------------------------------------------------
__global__ void __launch_bounds__(256)
knn_solve_kernel(const float* __restrict__ input,
                 const float* __restrict__ target,
                 const float* __restrict__ dbX,
                 const float* __restrict__ dbY,
                 const unsigned* __restrict__ candKey,
                 const int* __restrict__ candIdx,
                 float* __restrict__ out)
{
  extern __shared__ char smem[];
  unsigned* keys = (unsigned*)(smem);                             // CPR
  int* kidx      = (int*)(smem + (size_t)CPR * 4);                // CPR
  int* idxSel    = (int*)(smem + (size_t)CPR * 8);                // KNN
  float* Amat    = (float*)(smem + (size_t)CPR * 8 + KNN * 4);    // KNN*PDIM
  float* yv      = Amat + KNN * PDIM;                             // KNN
  float* Msys    = yv + KNN;                                      // PDIM*(PDIM+1)
  float* beta    = Msys + PDIM * (PDIM + 1);                      // PDIM
  int*   cntp    = (int*)(beta + PDIM);                           // 1

  const int b   = blockIdx.x;
  const int tid = threadIdx.x;

  for (int i = tid; i < CPR; i += 256) {
    keys[i] = candKey[(size_t)b * CPR + i];
    kidx[i] = candIdx[(size_t)b * CPR + i];
  }
  __syncthreads();

  // Block-wide bit bisection for the K-th smallest key among 4900 candidates.
  unsigned prefix = 0;
  for (int bit = 31; bit >= 0; --bit) {
    const unsigned trial = prefix | (1u << bit);
    if (tid == 0) *cntp = 0;
    __syncthreads();
    int c = 0;
    for (int i = tid; i < CPR; i += 256) c += (keys[i] < trial) ? 1 : 0;
    atomicAdd(cntp, c);
    __syncthreads();
    if (*cntp <= KNN - 1) prefix = trial;
    __syncthreads();
  }
  const unsigned T = prefix;
  if (tid == 0) *cntp = 0;
  __syncthreads();
  for (int i = tid; i < CPR; i += 256)
    if (keys[i] < T) { const int p = atomicAdd(cntp, 1); idxSel[p] = kidx[i]; }
  __syncthreads();
  for (int i = tid; i < CPR; i += 256)
    if (keys[i] == T) { const int p = atomicAdd(cntp, 1); if (p < KNN) idxSel[p] = kidx[i]; }
  __syncthreads();

  // Gather design matrix A = [1, Xn] and y.
  for (int i = tid; i < KNN * PDIM; i += 256) {
    const int j = i / PDIM, p = i % PDIM;
    Amat[i] = (p == 0) ? 1.0f : dbX[(size_t)idxSel[j] * DIM + (p - 1)];
  }
  for (int i = tid; i < KNN; i += 256) yv[i] = dbY[idxSel[i]];
  __syncthreads();

  // Normal equations, augmented: Msys = [AtA + jitter*I | Aty].
  for (int i = tid; i < PDIM * (PDIM + 1); i += 256) {
    const int p = i / (PDIM + 1), q = i % (PDIM + 1);
    float s = 0.f;
    if (q < PDIM) {
      for (int j = 0; j < KNN; ++j) s += Amat[j * PDIM + p] * Amat[j * PDIM + q];
      if (p == q) s += JITTER;
    } else {
      for (int j = 0; j < KNN; ++j) s += Amat[j * PDIM + p] * yv[j];
    }
    Msys[i] = s;
  }
  __syncthreads();

  // Forward elimination (SPD + jitter -> no pivoting needed).
  for (int col = 0; col < PDIM; ++col) {
    const int r = tid;
    if (r > col && r < PDIM) {
      const float f = Msys[r * (PDIM + 1) + col] / Msys[col * (PDIM + 1) + col];
      for (int c = col; c <= PDIM; ++c)
        Msys[r * (PDIM + 1) + c] -= f * Msys[col * (PDIM + 1) + c];
    }
    __syncthreads();
  }

  if (tid == 0) {
    for (int c = PDIM - 1; c >= 0; --c) {
      float s = Msys[c * (PDIM + 1) + PDIM];
      for (int q = c + 1; q < PDIM; ++q) s -= Msys[c * (PDIM + 1) + q] * beta[q];
      beta[c] = s / Msys[c * (PDIM + 1) + c];
    }
    float pred = beta[0];
    for (int d = 0; d < DIM; ++d) pred += input[(size_t)b * DIM + d] * beta[1 + d];
    const float diff = pred - target[b];
    atomicAdd(out, diff * diff * (1.0f / (float)BQ));
  }
}

__global__ void zero_out_kernel(float* out) {
  if (threadIdx.x == 0 && blockIdx.x == 0) out[0] = 0.0f;
}

extern "C" void kernel_launch(void* const* d_in, const int* in_sizes, int n_in,
                              void* d_out, int out_size, void* d_ws, size_t ws_size,
                              hipStream_t stream) {
  (void)in_sizes; (void)n_in; (void)out_size; (void)ws_size;

  const float* input  = (const float*)d_in[0];   // [1024, 64]
  const float* target = (const float*)d_in[1];   // [1024, 1]
  const float* dbX    = (const float*)d_in[2];   // [100000, 64]
  const float* dbY    = (const float*)d_in[3];   // [100000, 1]
  float* out = (float*)d_out;

  // Workspace layout
  char* ws = (char*)d_ws;
  _Float16* dbXh    = (_Float16*)ws;                               // NPAD*64 f16
  float*    d2      = (float*)(ws + (size_t)NPAD * DIM * 2);       // NPAD f32
  unsigned* candKey = (unsigned*)(ws + (size_t)NPAD * DIM * 2 + (size_t)NPAD * 4);
  int*      candIdx = (int*)((char*)candKey + (size_t)BQ * CPR * 4);

  zero_out_kernel<<<1, 32, 0, stream>>>(out);

  prep_db_kernel<<<NPAD / 256, 256, 0, stream>>>(dbX, dbXh, d2);

  dim3 gridA(NCHUNK, BQ / 16);
  const size_t ldsA = 198784;   // see layout comment in kernel A
  dist_topk_chunk_kernel<<<gridA, 256, ldsA, stream>>>(input, dbXh, d2,
                                                       candKey, candIdx);

  const size_t ldsB = (size_t)CPR * 8 + KNN * 4 + (size_t)KNN * PDIM * 4 +
                      KNN * 4 + (size_t)PDIM * (PDIM + 1) * 4 + PDIM * 4 + 16;
  knn_solve_kernel<<<BQ, 256, ldsB, stream>>>(input, target, dbX, dbY,
                                              candKey, candIdx, out);
}
